// LanguageModel_46153718563306
// MI455X (gfx1250) — matrix operand order repaired and checked by reading, not compiled, not run on hardware
//
#include <hip/hip_runtime.h>
#include <hip/hip_bf16.h>

typedef __attribute__((ext_vector_type(16))) __bf16 bf16x16;
typedef __attribute__((ext_vector_type(8)))  float  f32x8;
typedef __attribute__((ext_vector_type(4)))  float  f32x4;

constexpr int B_ = 2, N_ = 1024, D_ = 1024, V_ = 32000, L_ = 4, E_ = 4, F_ = 4096;
constexpr int TOK = B_ * N_;          // 2048 tokens
constexpr int BM = 64;                // rows per GEMM block (4 row tiles)
constexpr int BN = 256;               // cols per GEMM block (8 waves x 2 tiles x 16)
constexpr int RT = 4;                 // row tiles per block
constexpr int NT = 2;                 // col tiles per wave
constexpr int ROWS_PAD = 2304;        // expert regions padded to 64 rows (36 blocks)
constexpr int ROWBLK = ROWS_PAD / BM; // 36
constexpr int APITCH = 40;            // LDS A pitch in bf16 (80B, 8B-aligned)

// ---------------------------------------------------------------- utilities
__device__ inline float block_reduce_sum(float v, float* red) {
  int tid = threadIdx.x;
  red[tid] = v;
  __syncthreads();
  for (int s = 128; s > 0; s >>= 1) {
    if (tid < s) red[tid] += red[tid + s];
    __syncthreads();
  }
  float r = red[0];
  __syncthreads();
  return r;
}

__device__ inline float gelu_f(float x) {
  const float c0 = 0.7978845608028654f;  // sqrt(2/pi)
  float t = tanhf(c0 * (x + 0.044715f * x * x * x));
  return 0.5f * x * (1.0f + t);
}

// A fragment (16x32 bf16) gather per CDNA5 ISA layout; As points at a 16-row tile.
__device__ inline bf16x16 gather_a(const __bf16 (*As)[APITCH], int lm, int lhi) {
  bf16x16 af;
#pragma unroll
  for (int j = 0; j < 16; ++j) {
    int ka = (j < 8) ? (8 * lhi + j) : (8 + 8 * lhi + j);  // == 16+8*hi+(j-8)
    af[j] = As[lm][ka];
  }
  return af;
}

// B fragment (32x16) from global row-major [K x ldb], cvt f32->bf16 in regs.
__device__ inline bf16x16 load_b(const float* __restrict__ Wp, size_t ldb,
                                 int kc, int lhi, int ncol) {
  bf16x16 bf_;
#pragma unroll
  for (int j = 0; j < 16; ++j) {
    int kb = (j < 8) ? (8 * lhi + j) : (16 + 8 * lhi + (j - 8));
    bf_[j] = (__bf16)Wp[(size_t)(kc + kb) * ldb + ncol];
  }
  return bf_;
}

// ---------------------------------------------------------------- embedding
__global__ void embed_kernel(const int* __restrict__ ids,
                             const float* __restrict__ te,
                             const float* __restrict__ pe,
                             float* __restrict__ x) {
  int bn = blockIdx.x;
  int n = bn % N_;
  int id = ids[bn];
  for (int d = threadIdx.x; d < D_; d += 256)
    x[(size_t)bn * D_ + d] = te[(size_t)id * D_ + d] + pe[(size_t)n * D_ + d];
}

// -------------------------------------------------- LayerNorm (+ fused v-dot)
__global__ void ln_kernel(const float* __restrict__ x,
                          const float* __restrict__ w, const float* __restrict__ b,
                          const float* __restrict__ Wv, const float* __restrict__ bv,
                          float* __restrict__ y, float* __restrict__ vout,
                          int l, int has_v) {
  __shared__ float red[256];
  int t = blockIdx.x;
  const float* xt = x + (size_t)t * D_;
  float s = 0.f, s2 = 0.f;
  for (int d = threadIdx.x; d < D_; d += 256) {
    float xv = xt[d];
    s += xv; s2 += xv * xv;
  }
  float mu = block_reduce_sum(s, red) * (1.0f / D_);
  float ex2 = block_reduce_sum(s2, red) * (1.0f / D_);
  float inv = rsqrtf(ex2 - mu * mu + 1e-5f);
  float vdot = 0.f;
  for (int d = threadIdx.x; d < D_; d += 256) {
    float yv = (xt[d] - mu) * inv * w[l * D_ + d] + b[l * D_ + d];
    y[(size_t)t * D_ + d] = yv;
    if (has_v) vdot += yv * Wv[l * D_ + d];
  }
  if (has_v) {
    float vt = block_reduce_sum(vdot, red);
    if (threadIdx.x == 0)
      vout[t] = fminf(fmaxf(vt + bv[l], -3.0f), 3.0f);
  }
}

// -------------------------------- two-sided Thomas scans + diagonal Green fn
__global__ void scan_kernel(const float* __restrict__ v,
                            float* __restrict__ dr, float* __restrict__ di,
                            float* __restrict__ er, float* __restrict__ ei,
                            float* __restrict__ feats) {
  int tid = threadIdx.x;
  if (tid < 2 * B_) {
    int bb = tid >> 1, dir = tid & 1;
    const float* vb = v + bb * N_;
    if (dir == 0) {  // forward: d_i = a_i - 1/d_{i-1}
      float cr = -2.f + vb[0], ci = -1.f;
      dr[bb * N_] = cr; di[bb * N_] = ci;
      for (int i = 1; i < N_; ++i) {
        float inv = 1.f / (cr * cr + ci * ci);
        cr = (-2.f + vb[i]) - cr * inv;
        ci = -1.f + ci * inv;
        dr[bb * N_ + i] = cr; di[bb * N_ + i] = ci;
      }
    } else {         // backward: e_i = a_i - 1/e_{i+1}
      float cr = -2.f + vb[N_ - 1], ci = -1.f;
      er[bb * N_ + N_ - 1] = cr; ei[bb * N_ + N_ - 1] = ci;
      for (int i = N_ - 2; i >= 0; --i) {
        float inv = 1.f / (cr * cr + ci * ci);
        cr = (-2.f + vb[i]) - cr * inv;
        ci = -1.f + ci * inv;
        er[bb * N_ + i] = cr; ei[bb * N_ + i] = ci;
      }
    }
  }
  __syncthreads();
  for (int i = tid; i < TOK; i += blockDim.x) {
    float ar = -2.f + v[i], ai = -1.f;
    float xr = dr[i] + er[i] - ar, xi = di[i] + ei[i] - ai;
    float inv = 1.f / (xr * xr + xi * xi);
    float gr = xr * inv, gi = -xi * inv;
    feats[2 * i + 0] = fminf(fmaxf(gr, -10.f), 10.f);
    feats[2 * i + 1] = fminf(fmaxf(gi, -10.f), 10.f);
  }
}

// ---------------------------------------------------------------- routing
__global__ void route_init_kernel(int* __restrict__ cnt, int* __restrict__ perm) {
  int i = blockIdx.x * 256 + threadIdx.x;
  if (i < 8) cnt[i] = 0;
  if (i < ROWS_PAD) perm[i] = -1;
}

__global__ void gate_kernel(const float* __restrict__ y,
                            const float* __restrict__ Wg, const float* __restrict__ bg,
                            float* __restrict__ wgt, int* __restrict__ idxb,
                            int* __restrict__ cnt, int l) {
  __shared__ float red[256];
  int t = blockIdx.x;
  const float* yt = y + (size_t)t * D_;
  const float* Wl = Wg + (size_t)l * D_ * E_;
  float a0 = 0, a1 = 0, a2 = 0, a3 = 0;
  for (int d = threadIdx.x; d < D_; d += 256) {
    float yd = yt[d];
    a0 += yd * Wl[d * 4 + 0]; a1 += yd * Wl[d * 4 + 1];
    a2 += yd * Wl[d * 4 + 2]; a3 += yd * Wl[d * 4 + 3];
  }
  float g0 = block_reduce_sum(a0, red);
  float g1 = block_reduce_sum(a1, red);
  float g2 = block_reduce_sum(a2, red);
  float g3 = block_reduce_sum(a3, red);
  if (threadIdx.x == 0) {
    float g[4] = {g0 + bg[l * 4 + 0], g1 + bg[l * 4 + 1],
                  g2 + bg[l * 4 + 2], g3 + bg[l * 4 + 3]};
    float mx = fmaxf(fmaxf(g[0], g[1]), fmaxf(g[2], g[3]));
    float p[4], s = 0.f;
    for (int e = 0; e < 4; ++e) { p[e] = expf(g[e] - mx); s += p[e]; }
    float pm = -1.f; int am = 0;
    for (int e = 0; e < 4; ++e) {
      float pe = p[e] / s;
      if (pe > pm) { pm = pe; am = e; }
    }
    wgt[t] = pm;
    idxb[t] = am;
    atomicAdd(&cnt[am], 1);
  }
}

__global__ void offsets_kernel(const int* __restrict__ cnt, int* __restrict__ cursor,
                               int* __restrict__ rexp) {
  __shared__ int s[5];
  if (threadIdx.x == 0) {
    int off = 0;
    for (int e = 0; e < 4; ++e) {
      s[e] = off;
      cursor[e] = off;
      off += cnt[e];
      off = (off + BM - 1) & ~(BM - 1);  // pad each expert region to 64-row blocks
    }
    s[4] = off;
  }
  __syncthreads();
  for (int r = threadIdx.x; r < ROWS_PAD; r += 256) {
    int e = 3;
    for (int q = 0; q < 4; ++q)
      if (r < s[q + 1]) { e = q; break; }
    rexp[r] = e;
  }
}

__global__ void scatter_kernel(const int* __restrict__ idxb, int* __restrict__ cursor,
                               int* __restrict__ perm) {
  int t = blockIdx.x * 256 + threadIdx.x;
  if (t < TOK) {
    int e = idxb[t];
    int p = atomicAdd(&cursor[e], 1);
    perm[p] = t;
  }
}

// ------------------------------------------- MoE FFN GEMM1: h = gelu(y*W1+b1)
// 256 threads = 8 waves; block tile 64x256; wave: 4 row tiles x 2 col tiles.
__global__ void __launch_bounds__(256)
moe_ffn1_kernel(const float* __restrict__ y,
                const float* __restrict__ W1, const float* __restrict__ b1,
                const int* __restrict__ perm, const int* __restrict__ rexp,
                __bf16* __restrict__ h, int l) {
  __shared__ __bf16 As[BM][APITCH];
  __shared__ int toks[BM];
  int m0 = blockIdx.y * BM;
  int n0 = blockIdx.x * BN;
  int tid = threadIdx.x, wave = tid >> 5, lane = tid & 31;
  int lm = lane & 15, lhi = lane >> 4;
  int nw = n0 + wave * (NT * 16);
  int e = rexp[m0];
  const float* Wp = W1 + (size_t)(l * E_ + e) * D_ * F_;
  if (tid < BM) toks[tid] = perm[m0 + tid];
  f32x8 acc[RT][NT] = {};
  for (int kc = 0; kc < D_; kc += 32) {
    __syncthreads();
#pragma unroll
    for (int p = 0; p < 2; ++p) {  // stage 64x32 f32->bf16: 2 float4/thread
      int q = tid + p * 256;
      int r = q >> 3, kk = (q & 7) * 4;
      int t = toks[r];
      float msk = (t < 0) ? 0.f : 1.f;
      int ts = (t < 0) ? 0 : t;
      f32x4 v4 = *(const f32x4*)&y[(size_t)ts * D_ + kc + kk];
      As[r][kk + 0] = (__bf16)(v4.x * msk);
      As[r][kk + 1] = (__bf16)(v4.y * msk);
      As[r][kk + 2] = (__bf16)(v4.z * msk);
      As[r][kk + 3] = (__bf16)(v4.w * msk);
    }
    __syncthreads();
    bf16x16 af[RT];
#pragma unroll
    for (int rt = 0; rt < RT; ++rt) af[rt] = gather_a(As + rt * 16, lm, lhi);
#pragma unroll
    for (int tt = 0; tt < NT; ++tt) {
      bf16x16 bf_ = load_b(Wp, F_, kc, lhi, nw + tt * 16 + lm);
#pragma unroll
      for (int rt = 0; rt < RT; ++rt)
        acc[rt][tt] = __builtin_amdgcn_wmma_f32_16x16x32_bf16(
            false, af[rt], false, bf_, (short)0, acc[rt][tt], false, false);
    }
  }
#pragma unroll
  for (int tt = 0; tt < NT; ++tt) {
    int n = nw + tt * 16 + lm;
    float bb = b1[(size_t)(l * E_ + e) * F_ + n];
#pragma unroll
    for (int rt = 0; rt < RT; ++rt) {
#pragma unroll
      for (int r = 0; r < 8; ++r) {
        int m = rt * 16 + r + 8 * lhi;
        if (toks[m] >= 0)
          h[(size_t)(m0 + m) * F_ + n] = (__bf16)gelu_f(acc[rt][tt][r] + bb);
      }
    }
  }
}

// ------------------------- MoE FFN GEMM2: x[t] += w[t]*(h*W2 + b2), scattered
// A (h) is already bf16: stage via CDNA5 async global->LDS DMA path.
__global__ void __launch_bounds__(256)
moe_ffn2_kernel(const __bf16* __restrict__ h,
                const float* __restrict__ W2, const float* __restrict__ b2,
                const int* __restrict__ perm, const int* __restrict__ rexp,
                const float* __restrict__ wgt,
                float* __restrict__ x, int l) {
  __shared__ __bf16 As[BM][APITCH];
  __shared__ int toks[BM];
  int m0 = blockIdx.y * BM;
  int n0 = blockIdx.x * BN;
  int tid = threadIdx.x, wave = tid >> 5, lane = tid & 31;
  int lm = lane & 15, lhi = lane >> 4;
  int nw = n0 + wave * (NT * 16);
  int e = rexp[m0];
  const float* Wp = W2 + (size_t)(l * E_ + e) * F_ * D_;
  if (tid < BM) toks[tid] = perm[m0 + tid];
  f32x8 acc[RT][NT] = {};
  for (int kc = 0; kc < F_; kc += 32) {
    __syncthreads();
#pragma unroll
    for (int p = 0; p < 2; ++p) {  // async copy 64x32 bf16 (8B per lane-op)
      int q = tid + p * 256;
      int r = q >> 3, kk = (q & 7) * 4;
      unsigned lds = (unsigned)(uintptr_t)&As[r][kk];
      unsigned long long ga =
          (unsigned long long)(uintptr_t)&h[(size_t)(m0 + r) * F_ + kc + kk];
      asm volatile("global_load_async_to_lds_b64 %0, %1, off"
                   :: "v"(lds), "v"(ga) : "memory");
    }
    asm volatile("s_wait_asynccnt 0x0" ::: "memory");
    __syncthreads();
    bf16x16 af[RT];
#pragma unroll
    for (int rt = 0; rt < RT; ++rt) af[rt] = gather_a(As + rt * 16, lm, lhi);
#pragma unroll
    for (int tt = 0; tt < NT; ++tt) {
      bf16x16 bf_ = load_b(Wp, D_, kc, lhi, nw + tt * 16 + lm);
#pragma unroll
      for (int rt = 0; rt < RT; ++rt)
        acc[rt][tt] = __builtin_amdgcn_wmma_f32_16x16x32_bf16(
            false, af[rt], false, bf_, (short)0, acc[rt][tt], false, false);
    }
  }
#pragma unroll
  for (int tt = 0; tt < NT; ++tt) {
    int n = nw + tt * 16 + lm;
    float bb = b2[(size_t)(l * E_ + e) * D_ + n];
#pragma unroll
    for (int rt = 0; rt < RT; ++rt) {
#pragma unroll
      for (int r = 0; r < 8; ++r) {
        int m = rt * 16 + r + 8 * lhi;
        int t = toks[m];
        if (t >= 0)
          x[(size_t)t * D_ + n] += wgt[t] * (acc[rt][tt][r] + bb);
      }
    }
  }
}

// ------------------------------------------- spectral residual: x += bk*spec
__global__ void spec_add_kernel(float* __restrict__ x, const float* __restrict__ feats,
                                const float* __restrict__ Wo, const float* __restrict__ bo,
                                const float* __restrict__ bk, int l) {
  int t = blockIdx.x;
  float f0 = feats[2 * t + 0], f1 = feats[2 * t + 1];
  for (int d = threadIdx.x; d < D_; d += 256) {
    float spec = f0 * Wo[(size_t)(l * 2 + 0) * D_ + d] +
                 f1 * Wo[(size_t)(l * 2 + 1) * D_ + d] + bo[l * D_ + d];
    x[(size_t)t * D_ + d] += bk[l * D_ + d] * spec;
  }
}

// ------------------------------------------------------- LM head WMMA GEMM
__global__ void __launch_bounds__(256)
lm_gemm_kernel(const float* __restrict__ Xln,
               const float* __restrict__ Wlm,
               const float* __restrict__ blm,
               float* __restrict__ out) {
  __shared__ __bf16 As[BM][APITCH];
  int m0 = blockIdx.y * BM;
  int n0 = blockIdx.x * BN;
  int tid = threadIdx.x, wave = tid >> 5, lane = tid & 31;
  int lm = lane & 15, lhi = lane >> 4;
  int nw = n0 + wave * (NT * 16);
  f32x8 acc[RT][NT] = {};
  for (int kc = 0; kc < D_; kc += 32) {
    __syncthreads();
#pragma unroll
    for (int p = 0; p < 2; ++p) {
      int q = tid + p * 256;
      int r = q >> 3, kk = (q & 7) * 4;
      f32x4 v4 = *(const f32x4*)&Xln[(size_t)(m0 + r) * D_ + kc + kk];
      As[r][kk + 0] = (__bf16)v4.x;
      As[r][kk + 1] = (__bf16)v4.y;
      As[r][kk + 2] = (__bf16)v4.z;
      As[r][kk + 3] = (__bf16)v4.w;
    }
    __syncthreads();
    bf16x16 af[RT];
#pragma unroll
    for (int rt = 0; rt < RT; ++rt) af[rt] = gather_a(As + rt * 16, lm, lhi);
#pragma unroll
    for (int tt = 0; tt < NT; ++tt) {
      bf16x16 bf_ = load_b(Wlm, V_, kc, lhi, nw + tt * 16 + lm);
#pragma unroll
      for (int rt = 0; rt < RT; ++rt)
        acc[rt][tt] = __builtin_amdgcn_wmma_f32_16x16x32_bf16(
            false, af[rt], false, bf_, (short)0, acc[rt][tt], false, false);
    }
  }
#pragma unroll
  for (int tt = 0; tt < NT; ++tt) {
    int n = nw + tt * 16 + lm;
    float bn = blm[n];
#pragma unroll
    for (int rt = 0; rt < RT; ++rt) {
#pragma unroll
      for (int r = 0; r < 8; ++r) {
        int m = m0 + rt * 16 + r + 8 * lhi;
        out[(size_t)m * V_ + n] = acc[rt][tt][r] + bn;
      }
    }
  }
}

// ---------------------------------------------------------------- host side
extern "C" void kernel_launch(void* const* d_in, const int* in_sizes, int n_in,
                              void* d_out, int out_size, void* d_ws, size_t ws_size,
                              hipStream_t stream) {
  const int*   token_ids = (const int*)  d_in[0];
  const float* tok_emb   = (const float*)d_in[1];
  const float* pos_emb   = (const float*)d_in[2];
  const float* ln_w      = (const float*)d_in[3];
  const float* ln_b      = (const float*)d_in[4];
  const float* Wg        = (const float*)d_in[5];
  const float* bg        = (const float*)d_in[6];
  const float* W1        = (const float*)d_in[7];
  const float* b1        = (const float*)d_in[8];
  const float* W2        = (const float*)d_in[9];
  const float* b2        = (const float*)d_in[10];
  const float* Wv        = (const float*)d_in[11];
  const float* bv        = (const float*)d_in[12];
  const float* Wo        = (const float*)d_in[13];
  const float* bo        = (const float*)d_in[14];
  const float* bk_scale  = (const float*)d_in[15];
  const float* lnf_w     = (const float*)d_in[16];
  const float* lnf_b     = (const float*)d_in[17];
  const float* Wlm       = (const float*)d_in[18];
  const float* blm       = (const float*)d_in[19];
  float* out = (float*)d_out;

  char* base = (char*)d_ws;
  auto alloc = [&](size_t bytes) {
    void* p = (void*)base;
    base += (bytes + 255) & ~(size_t)255;
    return p;
  };
  float*  x     = (float*)alloc((size_t)TOK * D_ * 4);
  float*  y     = (float*)alloc((size_t)TOK * D_ * 4);
  float*  vbuf  = (float*)alloc(TOK * 4);
  float*  dr    = (float*)alloc(TOK * 4);
  float*  di    = (float*)alloc(TOK * 4);
  float*  er    = (float*)alloc(TOK * 4);
  float*  ei    = (float*)alloc(TOK * 4);
  float*  feats = (float*)alloc(TOK * 2 * 4);
  float*  wgt   = (float*)alloc(TOK * 4);
  int*    idxb  = (int*)  alloc(TOK * 4);
  int*    cnt   = (int*)  alloc(16 * 4);
  int*    cursor= (int*)  alloc(16 * 4);
  int*    perm  = (int*)  alloc(ROWS_PAD * 4);
  int*    rexp  = (int*)  alloc(ROWS_PAD * 4);
  __bf16* h     = (__bf16*)alloc((size_t)ROWS_PAD * F_ * 2);

  embed_kernel<<<TOK, 256, 0, stream>>>(token_ids, tok_emb, pos_emb, x);

  for (int l = 0; l < L_; ++l) {
    ln_kernel<<<TOK, 256, 0, stream>>>(x, ln_w, ln_b, Wv, bv, y, vbuf, l, 1);
    scan_kernel<<<1, 256, 0, stream>>>(vbuf, dr, di, er, ei, feats);
    route_init_kernel<<<(ROWS_PAD + 255) / 256, 256, 0, stream>>>(cnt, perm);
    gate_kernel<<<TOK, 256, 0, stream>>>(y, Wg, bg, wgt, idxb, cnt, l);
    offsets_kernel<<<1, 256, 0, stream>>>(cnt, cursor, rexp);
    scatter_kernel<<<(TOK + 255) / 256, 256, 0, stream>>>(idxb, cursor, perm);
    moe_ffn1_kernel<<<dim3(F_ / BN, ROWBLK), 256, 0, stream>>>(y, W1, b1, perm, rexp, h, l);
    moe_ffn2_kernel<<<dim3(D_ / BN, ROWBLK), 256, 0, stream>>>(h, W2, b2, perm, rexp, wgt, x, l);
    spec_add_kernel<<<TOK, 256, 0, stream>>>(x, feats, Wo, bo, bk_scale, l);
  }

  ln_kernel<<<TOK, 256, 0, stream>>>(x, lnf_w, lnf_b, nullptr, nullptr, y, nullptr, 0, 0);
  lm_gemm_kernel<<<dim3(V_ / BN, TOK / BM), 256, 0, stream>>>(y, Wlm, blm, out);

  (void)in_sizes; (void)n_in; (void)out_size; (void)ws_size;
}
